// DualPathTransformer_10814727651586
// MI455X (gfx1250) — matrix operand order repaired and checked by reading, not compiled, not run on hardware
//
#include <hip/hip_runtime.h>
#include <hip/hip_bf16.h>
#include <stdint.h>

#define DEV __device__ __forceinline__

typedef unsigned short u16;
typedef __attribute__((ext_vector_type(16))) __bf16 bf16x16;
typedef __attribute__((ext_vector_type(8)))  float  f32x8;
typedef __attribute__((ext_vector_type(8)))  unsigned int u32x8;
typedef __attribute__((ext_vector_type(4)))  unsigned int u32x4;

union BF16Frag { u32x8 u; bf16x16 b; };

// Problem dims (compile-time constants)
constexpr int B_ = 4, S_ = 2048, DIN_ = 256, D_ = 512, H_ = 8, HD_ = 64, DOUT_ = 128;
constexpr int M_ = B_ * S_;           // 8192 rows
constexpr float SCALE_ = 0.125f;      // 1/sqrt(HD)
constexpr int HW_ = 32;               // W/2 band half-width

DEV u16 f2bf(float f) {
  union { float f; unsigned u; } x; x.f = f;
  unsigned r = x.u + 0x7fffu + ((x.u >> 16) & 1u);  // round-to-nearest-even
  return (u16)(r >> 16);
}

DEV f32x8 wmma_bf16(const bf16x16 a, const bf16x16 b, const f32x8 c) {
  // D = A(16x32 bf16) * B(32x16 bf16) + C(16x16 f32)
  return __builtin_amdgcn_wmma_f32_16x16x32_bf16(false, a, false, b, (short)0, c, false, false);
}

// A-operand (16x32): lanes 0-15 hold row=lm, K {0..15}; lanes 16-31 row=lm, K {16..31}
DEV void load_a_frag(BF16Frag& f, const u16* row, int kb, int lh) {
  u32x4 lo = *(const u32x4*)(row + kb + (lh ? 16 : 0));
  u32x4 hi = *(const u32x4*)(row + kb + 8 + (lh ? 16 : 0));
  f.u[0] = lo[0]; f.u[1] = lo[1]; f.u[2] = lo[2]; f.u[3] = lo[3];
  f.u[4] = hi[0]; f.u[5] = hi[1]; f.u[6] = hi[2]; f.u[7] = hi[3];
}

// B-operand (32x16): lane holds one column; 16 contiguous bf16 (K slice) = 32 bytes
DEV void load_b_frag(BF16Frag& f, const u16* p) {
  u32x4 lo = *(const u32x4*)(p);
  u32x4 hi = *(const u32x4*)(p + 8);
  f.u[0] = lo[0]; f.u[1] = lo[1]; f.u[2] = lo[2]; f.u[3] = lo[3];
  f.u[4] = hi[0]; f.u[5] = hi[1]; f.u[6] = hi[2]; f.u[7] = hi[3];
}

// ---- generic bf16 WMMA GEMM: 32(M)x64(N) tile per wave, double-buffered K pipeline ----
enum { EPI_HPOS = 0, EPI_HEADS, EPI_HEADST, EPI_F32BF, EPI_GATE, EPI_RELUBF, EPI_F32 };

template<int EPI>
__global__ __launch_bounds__(32) void gemm_wmma(
    const u16* __restrict__ A, const u16* __restrict__ Bt,
    const float* __restrict__ bias, int N, int K,
    float* __restrict__ Cf, u16* __restrict__ Cb, int cb_ld, int cb_off,
    const float* __restrict__ X0, const float* __restrict__ X1)
{
  const int lane = threadIdx.x;
  const int lh = lane >> 4, lm = lane & 15;
  const int tn = blockIdx.x, tm = blockIdx.y;
  const u16* ar0 = A + (size_t)(tm * 32 + lm) * K;
  const u16* ar1 = ar0 + (size_t)16 * K;
  const u16* br  = Bt + (size_t)(tn * 64 + lm) * K + lh * 16;
  const size_t bstep = (size_t)16 * K;

  f32x8 acc[2][4];
#pragma unroll
  for (int mt = 0; mt < 2; mt++)
#pragma unroll
    for (int t = 0; t < 4; t++)
#pragma unroll
      for (int i = 0; i < 8; i++) acc[mt][t][i] = 0.f;

  BF16Frag fa[2][2], fb[2][4];
  auto load_set = [&](int buf, int kb) {
    load_a_frag(fa[buf][0], ar0, kb, lh);
    load_a_frag(fa[buf][1], ar1, kb, lh);
#pragma unroll
    for (int t = 0; t < 4; t++) load_b_frag(fb[buf][t], br + t * bstep + kb);
  };
  auto mma_set = [&](int buf) {
#pragma unroll
    for (int t = 0; t < 4; t++) {
      acc[0][t] = wmma_bf16(fa[buf][0].b, fb[buf][t].b, acc[0][t]);
      acc[1][t] = wmma_bf16(fa[buf][1].b, fb[buf][t].b, acc[1][t]);
    }
  };

  // software pipeline: loads for step i+1 issued before MMAs of step i consume step-i data
  load_set(0, 0);
  for (int kb = 0; kb < K; kb += 64) {     // K is a multiple of 64 for all GEMMs here
    load_set(1, kb + 32);
    mma_set(0);
    if (kb + 64 < K) load_set(0, kb + 64);
    mma_set(1);
  }

#pragma unroll
  for (int mt = 0; mt < 2; mt++)
#pragma unroll
    for (int t = 0; t < 4; t++)
#pragma unroll
      for (int r = 0; r < 8; r++) {
        const int row = tm * 32 + mt * 16 + r + 8 * lh;  // C layout: VGPR r = M r / r+8 per half
        const int col = tn * 64 + t * 16 + lm;           // N = lane%16 within 16-wide tile
        float v = acc[mt][t][r] + bias[col];
        if constexpr (EPI == EPI_HPOS) {
          v += X0[(size_t)(row & (S_ - 1)) * D_ + col];          // + pos[s, col]
          Cf[(size_t)row * D_ + col] = v;
          Cb[(size_t)row * D_ + col] = f2bf(v);
        } else if constexpr (EPI == EPI_HEADS) {
          int h = col >> 6, d = col & 63, b = row >> 11, s = row & (S_ - 1);
          Cb[(((size_t)(b * H_ + h)) * S_ + s) * HD_ + d] = f2bf(v);
        } else if constexpr (EPI == EPI_HEADST) {                // V stored transposed [B,H,HD,S]
          int h = col >> 6, d = col & 63, b = row >> 11, s = row & (S_ - 1);
          Cb[(((size_t)(b * H_ + h)) * HD_ + d) * S_ + s] = f2bf(v);
        } else if constexpr (EPI == EPI_F32BF) {
          Cf[(size_t)row * N + col] = v;
          Cb[(size_t)row * cb_ld + cb_off + col] = f2bf(v);      // into concat buffer half
        } else if constexpr (EPI == EPI_GATE) {
          float gt = tanhf(fmaxf(v, 0.f));
          size_t i = (size_t)row * D_ + col;
          Cf[i] = gt * X0[i] + (1.f - gt) * X1[i];               // g*local + (1-g)*global
        } else if constexpr (EPI == EPI_RELUBF) {
          Cb[(size_t)row * cb_ld + col] = f2bf(fmaxf(v, 0.f));
        } else {
          Cf[(size_t)row * N + col] = v;
        }
      }
}

// ---------------- flash-style attention: 1 wave per (b,h, 16-query tile) ----------------
template<bool LOCAL>
__global__ __launch_bounds__(32) void attn_wmma(
    const u16* __restrict__ Q, const u16* __restrict__ Kmat,
    const u16* __restrict__ Vt, u16* __restrict__ AO)
{
  const int lane = threadIdx.x;
  const int lh = lane >> 4, lm = lane & 15;
  const int qt = blockIdx.x, bh = blockIdx.y;
  const int q0 = qt * 16;
  __shared__ u16 Plds[16 * 32];

  const u16* qrow = Q + ((size_t)bh * S_ + q0 + lm) * HD_;
  BF16Frag qa[2];
  load_a_frag(qa[0], qrow, 0, lh);
  load_a_frag(qa[1], qrow, 32, lh);

  f32x8 acc[4];
#pragma unroll
  for (int t = 0; t < 4; t++)
#pragma unroll
    for (int i = 0; i < 8; i++) acc[t][i] = 0.f;
  float mrow[8], lrow[8];
#pragma unroll
  for (int r = 0; r < 8; r++) { mrow[r] = -1e30f; lrow[r] = 0.f; }

  int kb_lo = 0, kb_hi = S_ - 1;
  if (LOCAL) {
    int lo = q0 - HW_; if (lo < 0) lo = 0;
    kb_lo = lo & ~31;
    kb_hi = q0 + 15 + HW_; if (kb_hi > S_ - 1) kb_hi = S_ - 1;
  }

  for (int kb = kb_lo; kb <= kb_hi; kb += 32) {
    // issue K-operand loads, then V-operand loads: scores can start while V is in flight,
    // and the softmax VALU work below overlaps the V latency.
    BF16Frag kf[4], vf[4];
#pragma unroll
    for (int t = 0; t < 2; t++)
#pragma unroll
      for (int c = 0; c < 2; c++)
        load_b_frag(kf[t * 2 + c],
                    Kmat + ((size_t)bh * S_ + kb + t * 16 + lm) * HD_ + c * 32 + lh * 16);
#pragma unroll
    for (int t = 0; t < 4; t++)
      load_b_frag(vf[t], Vt + ((size_t)bh * HD_ + t * 16 + lm) * S_ + kb + lh * 16);
    // prefetch next key block (global_prefetch_b8): K rows + Vt rows
    if (kb + 32 <= kb_hi) {
      __builtin_prefetch(Kmat + ((size_t)bh * S_ + kb + 32 + lane) * HD_, 0, 3);
      __builtin_prefetch(Vt + ((size_t)bh * HD_ + lane * 2) * S_ + kb + 32, 0, 3);
    }

    f32x8 sc[2];
#pragma unroll
    for (int t = 0; t < 2; t++) {
#pragma unroll
      for (int i = 0; i < 8; i++) sc[t][i] = 0.f;
#pragma unroll
      for (int c = 0; c < 2; c++) sc[t] = wmma_bf16(qa[c].b, kf[t * 2 + c].b, sc[t]);
    }

    float pv[2][8];
#pragma unroll
    for (int t = 0; t < 2; t++)
#pragma unroll
      for (int r = 0; r < 8; r++) {
        float xv = sc[t][r] * SCALE_;
        if constexpr (LOCAL) {
          int rq = q0 + r + 8 * lh;
          int ck = kb + t * 16 + lm;
          int dd = rq - ck;
          if (dd < -HW_ || dd > HW_) xv = -1e30f;  // finite sentinel: EXEC stays uniform, no NaNs
        }
        pv[t][r] = xv;
      }
    // online softmax row statistics (rows live on fixed (VGPR r, half) -> 16-lane butterflies)
    float fac[8];
#pragma unroll
    for (int r = 0; r < 8; r++) {
      float m = fmaxf(pv[0][r], pv[1][r]);
      for (int o = 1; o < 16; o <<= 1) m = fmaxf(m, __shfl_xor(m, o, 32));
      float nm = fmaxf(mrow[r], m);
      fac[r] = __expf(mrow[r] - nm);               // -1e30-(-1e30)=0 -> 1 (harmless: acc=l=0)
      mrow[r] = nm;
      float p0 = __expf(pv[0][r] - nm);
      float p1 = __expf(pv[1][r] - nm);
      float rs = p0 + p1;
      for (int o = 1; o < 16; o <<= 1) rs += __shfl_xor(rs, o, 32);
      lrow[r] = lrow[r] * fac[r] + rs;
      pv[0][r] = p0; pv[1][r] = p1;
    }
    // transpose P (C-layout -> A-layout) through LDS, as bf16
#pragma unroll
    for (int t = 0; t < 2; t++)
#pragma unroll
      for (int r = 0; r < 8; r++)
        Plds[(r + 8 * lh) * 32 + t * 16 + lm] = f2bf(pv[t][r]);
    __syncthreads();
    BF16Frag pa;
    {
      const unsigned int* prow = (const unsigned int*)(Plds + lm * 32);
      const int j0 = lh ? 8 : 0;
#pragma unroll
      for (int i = 0; i < 4; i++) { pa.u[i] = prow[j0 + i]; pa.u[4 + i] = prow[4 + j0 + i]; }
    }
    __syncthreads();
    // acc = acc*fac + P(16x32) @ Vblock(32x64)   (V transposed: contiguous B loads, issued early)
#pragma unroll
    for (int t = 0; t < 4; t++) {
#pragma unroll
      for (int r = 0; r < 8; r++) acc[t][r] *= fac[r];
      acc[t] = wmma_bf16(pa.b, vf[t].b, acc[t]);
    }
  }
  const int b = bh >> 3, h = bh & 7;
#pragma unroll
  for (int t = 0; t < 4; t++)
#pragma unroll
    for (int r = 0; r < 8; r++) {
      float l = lrow[r];
      float v = acc[t][r] / (l > 0.f ? l : 1.f);
      size_t row = (size_t)b * S_ + q0 + r + 8 * lh;
      AO[row * D_ + h * 64 + t * 16 + lm] = f2bf(v);
    }
}

// ---------------- elementwise / LN / pool kernels ----------------
__global__ void cvt_bf16_kernel(const float* __restrict__ s, u16* __restrict__ d, int n) {
  int i = blockIdx.x * 256 + threadIdx.x;
  if (i < n) d[i] = f2bf(s[i]);
}
// wt[N][K] = bf16(w[K][N])  (transpose-convert weights once per launch)
__global__ void tconv_kernel(const float* __restrict__ w, u16* __restrict__ wt, int K, int N) {
  int i = blockIdx.x * 256 + threadIdx.x;
  if (i < K * N) { int n = i / K, k = i - n * K; wt[i] = f2bf(w[(size_t)k * N + n]); }
}

__global__ __launch_bounds__(256) void add_ln_kernel(
    const float* __restrict__ A, const float* __restrict__ R,
    const float* __restrict__ g, const float* __restrict__ be,
    float* __restrict__ outf, u16* __restrict__ outb)
{
  const int row = blockIdx.x, tid = threadIdx.x;
  const float* pa = A + (size_t)row * D_;
  const float* pr = R + (size_t)row * D_;
  float x0 = pa[tid] + pr[tid];
  float x1 = pa[tid + 256] + pr[tid + 256];
  __shared__ float r1[256], r2[256];
  r1[tid] = x0 + x1; r2[tid] = x0 * x0 + x1 * x1;
  __syncthreads();
  for (int o = 128; o > 0; o >>= 1) {
    if (tid < o) { r1[tid] += r1[tid + o]; r2[tid] += r2[tid + o]; }
    __syncthreads();
  }
  float mean = r1[0] * (1.f / D_);
  float var = fmaxf(r2[0] * (1.f / D_) - mean * mean, 0.f);
  float rstd = rsqrtf(var + 1e-5f);
  float y0 = (x0 - mean) * rstd * g[tid] + be[tid];
  float y1 = (x1 - mean) * rstd * g[tid + 256] + be[tid + 256];
  size_t o0 = (size_t)row * D_ + tid;
  outf[o0] = y0; outf[o0 + 256] = y1;
  outb[o0] = f2bf(y0); outb[o0 + 256] = f2bf(y1);
}

// h = LN(LN(A+R; n2); n3)
__global__ __launch_bounds__(256) void add_ln2_kernel(
    const float* __restrict__ A, const float* __restrict__ R,
    const float* __restrict__ g2, const float* __restrict__ b2,
    const float* __restrict__ g3, const float* __restrict__ b3,
    float* __restrict__ outf)
{
  const int row = blockIdx.x, tid = threadIdx.x;
  const float* pa = A + (size_t)row * D_;
  const float* pr = R + (size_t)row * D_;
  float x0 = pa[tid] + pr[tid];
  float x1 = pa[tid + 256] + pr[tid + 256];
  __shared__ float r1[256], r2[256];
  r1[tid] = x0 + x1; r2[tid] = x0 * x0 + x1 * x1;
  __syncthreads();
  for (int o = 128; o > 0; o >>= 1) {
    if (tid < o) { r1[tid] += r1[tid + o]; r2[tid] += r2[tid + o]; }
    __syncthreads();
  }
  float mean = r1[0] * (1.f / D_);
  float var = fmaxf(r2[0] * (1.f / D_) - mean * mean, 0.f);
  float rstd = rsqrtf(var + 1e-5f);
  float y0 = (x0 - mean) * rstd * g2[tid] + b2[tid];
  float y1 = (x1 - mean) * rstd * g2[tid + 256] + b2[tid + 256];
  __syncthreads();
  r1[tid] = y0 + y1; r2[tid] = y0 * y0 + y1 * y1;
  __syncthreads();
  for (int o = 128; o > 0; o >>= 1) {
    if (tid < o) { r1[tid] += r1[tid + o]; r2[tid] += r2[tid + o]; }
    __syncthreads();
  }
  float mean2 = r1[0] * (1.f / D_);
  float var2 = fmaxf(r2[0] * (1.f / D_) - mean2 * mean2, 0.f);
  float rstd2 = rsqrtf(var2 + 1e-5f);
  size_t o0 = (size_t)row * D_ + tid;
  outf[o0]       = (y0 - mean2) * rstd2 * g3[tid] + b3[tid];
  outf[o0 + 256] = (y1 - mean2) * rstd2 * g3[tid + 256] + b3[tid + 256];
}

__global__ __launch_bounds__(256) void pool_kernel(const float* __restrict__ h, float* __restrict__ pooled) {
  int idx = blockIdx.x * 256 + threadIdx.x;   // B_*D_ = 2048
  int b = idx >> 9, d = idx & 511;
  const float* p = h + ((size_t)b * S_) * D_ + d;
  float s = 0.f;
  for (int i = 0; i < S_; i++) s += p[(size_t)i * D_];
  pooled[idx] = s * (1.f / S_);
}

__global__ __launch_bounds__(256) void outproj_kernel(
    const float* __restrict__ pooled, const float* __restrict__ w,
    const float* __restrict__ bias, float* __restrict__ out)
{
  int idx = blockIdx.x * 256 + threadIdx.x;   // B_*DOUT_ = 512
  int b = idx >> 7, j = idx & 127;
  const float* p = pooled + b * D_;
  float s = bias[j];
  for (int d = 0; d < D_; d++) s += p[d] * w[(size_t)d * DOUT_ + j];
  out[idx] = s;
}

// ---------------- launch ----------------
extern "C" void kernel_launch(void* const* d_in, const int* in_sizes, int n_in,
                              void* d_out, int out_size, void* d_ws, size_t ws_size,
                              hipStream_t stream) {
  const float* x      = (const float*)d_in[0];
  const float* pos    = (const float*)d_in[1];
  const float* win_w  = (const float*)d_in[2];
  const float* win_b  = (const float*)d_in[3];
  const float* l_wqkv = (const float*)d_in[4];
  const float* l_bqkv = (const float*)d_in[5];
  const float* l_wo   = (const float*)d_in[6];
  const float* l_bo   = (const float*)d_in[7];
  const float* g_wqkv = (const float*)d_in[8];
  const float* g_bqkv = (const float*)d_in[9];
  const float* g_wo   = (const float*)d_in[10];
  const float* g_bo   = (const float*)d_in[11];
  const float* gate_w = (const float*)d_in[12];
  const float* gate_b = (const float*)d_in[13];
  const float* ffn_w1 = (const float*)d_in[14];
  const float* ffn_b1 = (const float*)d_in[15];
  const float* ffn_w2 = (const float*)d_in[16];
  const float* ffn_b2 = (const float*)d_in[17];
  const float* n1_g = (const float*)d_in[18]; const float* n1_b = (const float*)d_in[19];
  const float* n2_g = (const float*)d_in[20]; const float* n2_b = (const float*)d_in[21];
  const float* n3_g = (const float*)d_in[22]; const float* n3_b = (const float*)d_in[23];
  const float* out_w = (const float*)d_in[24]; const float* out_b = (const float*)d_in[25];

  char* ws = (char*)d_ws;
  size_t off = 0;
  auto alloc = [&](size_t bytes) { size_t o = off; off += (bytes + 255) & ~(size_t)255; return o; };
  const size_t MD2 = (size_t)M_ * D_ * 2, MD4 = (size_t)M_ * D_ * 4;

  u16* winT  = (u16*)(ws + alloc((size_t)D_ * DIN_ * 2));
  u16* lqT   = (u16*)(ws + alloc((size_t)D_ * D_ * 2));
  u16* lkT   = (u16*)(ws + alloc((size_t)D_ * D_ * 2));
  u16* lvT   = (u16*)(ws + alloc((size_t)D_ * D_ * 2));
  u16* gqT   = (u16*)(ws + alloc((size_t)D_ * D_ * 2));
  u16* gkT   = (u16*)(ws + alloc((size_t)D_ * D_ * 2));
  u16* gvT   = (u16*)(ws + alloc((size_t)D_ * D_ * 2));
  u16* lwoT  = (u16*)(ws + alloc((size_t)D_ * D_ * 2));
  u16* gwoT  = (u16*)(ws + alloc((size_t)D_ * D_ * 2));
  u16* gateT = (u16*)(ws + alloc((size_t)D_ * 2 * D_ * 2));
  u16* f1T   = (u16*)(ws + alloc((size_t)2 * D_ * D_ * 2));
  u16* f2T   = (u16*)(ws + alloc((size_t)D_ * 2 * D_ * 2));
  u16* xbf   = (u16*)(ws + alloc((size_t)M_ * DIN_ * 2));
  float* hf  = (float*)(ws + alloc(MD4));
  u16* hb    = (u16*)(ws + alloc(MD2));
  u16* qL    = (u16*)(ws + alloc(MD2));
  u16* kL    = (u16*)(ws + alloc(MD2));
  u16* vtL   = (u16*)(ws + alloc(MD2));
  u16* qG    = (u16*)(ws + alloc(MD2));
  u16* kG    = (u16*)(ws + alloc(MD2));
  u16* vtG   = (u16*)(ws + alloc(MD2));
  u16* aoL   = (u16*)(ws + alloc(MD2));
  u16* aoG   = (u16*)(ws + alloc(MD2));
  float* loF = (float*)(ws + alloc(MD4));
  float* goF = (float*)(ws + alloc(MD4));
  u16* catB  = (u16*)(ws + alloc((size_t)M_ * 2 * D_ * 2));
  float* fuF = (float*)(ws + alloc(MD4));
  float* h2F = (float*)(ws + alloc(MD4));
  u16* h2B   = (u16*)(ws + alloc(MD2));
  u16* midB  = (u16*)(ws + alloc((size_t)M_ * 2 * D_ * 2));
  float* ffF = (float*)(ws + alloc(MD4));
  float* h3F = (float*)(ws + alloc(MD4));
  float* plF = (float*)(ws + alloc((size_t)B_ * D_ * 4));
  (void)ws_size; (void)in_sizes; (void)n_in; (void)out_size;

  const dim3 b32(32), b256(256);
  const dim3 g512(D_ / 64, M_ / 32);       // N=512  -> (8, 256)
  const dim3 g1024(2 * D_ / 64, M_ / 32);  // N=1024 -> (16, 256)

  // 1) convert activations / weights to bf16 (weights transposed to [N][K])
  cvt_bf16_kernel<<<(M_ * DIN_ + 255) / 256, b256, 0, stream>>>(x, xbf, M_ * DIN_);
  tconv_kernel<<<(D_ * DIN_ + 255) / 256, b256, 0, stream>>>(win_w, winT, DIN_, D_);
  tconv_kernel<<<(D_ * D_ + 255) / 256, b256, 0, stream>>>(l_wqkv + 0 * D_ * D_, lqT, D_, D_);
  tconv_kernel<<<(D_ * D_ + 255) / 256, b256, 0, stream>>>(l_wqkv + 1 * D_ * D_, lkT, D_, D_);
  tconv_kernel<<<(D_ * D_ + 255) / 256, b256, 0, stream>>>(l_wqkv + 2 * D_ * D_, lvT, D_, D_);
  tconv_kernel<<<(D_ * D_ + 255) / 256, b256, 0, stream>>>(g_wqkv + 0 * D_ * D_, gqT, D_, D_);
  tconv_kernel<<<(D_ * D_ + 255) / 256, b256, 0, stream>>>(g_wqkv + 1 * D_ * D_, gkT, D_, D_);
  tconv_kernel<<<(D_ * D_ + 255) / 256, b256, 0, stream>>>(g_wqkv + 2 * D_ * D_, gvT, D_, D_);
  tconv_kernel<<<(D_ * D_ + 255) / 256, b256, 0, stream>>>(l_wo, lwoT, D_, D_);
  tconv_kernel<<<(D_ * D_ + 255) / 256, b256, 0, stream>>>(g_wo, gwoT, D_, D_);
  tconv_kernel<<<(2 * D_ * D_ + 255) / 256, b256, 0, stream>>>(gate_w, gateT, 2 * D_, D_);
  tconv_kernel<<<(2 * D_ * D_ + 255) / 256, b256, 0, stream>>>(ffn_w1, f1T, D_, 2 * D_);
  tconv_kernel<<<(2 * D_ * D_ + 255) / 256, b256, 0, stream>>>(ffn_w2, f2T, 2 * D_, D_);

  // 2) h = x @ win_w + b + pos
  gemm_wmma<EPI_HPOS><<<g512, b32, 0, stream>>>(xbf, winT, win_b, D_, DIN_, hf, hb, D_, 0, pos, nullptr);

  // 3) QKV for both paths (V written transposed)
  gemm_wmma<EPI_HEADS ><<<g512, b32, 0, stream>>>(hb, lqT, l_bqkv + 0 * D_, D_, D_, nullptr, qL, 0, 0, nullptr, nullptr);
  gemm_wmma<EPI_HEADS ><<<g512, b32, 0, stream>>>(hb, lkT, l_bqkv + 1 * D_, D_, D_, nullptr, kL, 0, 0, nullptr, nullptr);
  gemm_wmma<EPI_HEADST><<<g512, b32, 0, stream>>>(hb, lvT, l_bqkv + 2 * D_, D_, D_, nullptr, vtL, 0, 0, nullptr, nullptr);
  gemm_wmma<EPI_HEADS ><<<g512, b32, 0, stream>>>(hb, gqT, g_bqkv + 0 * D_, D_, D_, nullptr, qG, 0, 0, nullptr, nullptr);
  gemm_wmma<EPI_HEADS ><<<g512, b32, 0, stream>>>(hb, gkT, g_bqkv + 1 * D_, D_, D_, nullptr, kG, 0, 0, nullptr, nullptr);
  gemm_wmma<EPI_HEADST><<<g512, b32, 0, stream>>>(hb, gvT, g_bqkv + 2 * D_, D_, D_, nullptr, vtG, 0, 0, nullptr, nullptr);

  // 4) attention (banded local / full global), streaming softmax
  attn_wmma<true ><<<dim3(S_ / 16, B_ * H_), b32, 0, stream>>>(qL, kL, vtL, aoL);
  attn_wmma<false><<<dim3(S_ / 16, B_ * H_), b32, 0, stream>>>(qG, kG, vtG, aoG);

  // 5) output projections; bf16 copies land in concat buffer halves
  gemm_wmma<EPI_F32BF><<<g512, b32, 0, stream>>>(aoL, lwoT, l_bo, D_, D_, loF, catB, 2 * D_, 0, nullptr, nullptr);
  gemm_wmma<EPI_F32BF><<<g512, b32, 0, stream>>>(aoG, gwoT, g_bo, D_, D_, goF, catB, 2 * D_, D_, nullptr, nullptr);

  // 6) gate + fuse (fused epilogue)
  gemm_wmma<EPI_GATE><<<g512, b32, 0, stream>>>(catB, gateT, gate_b, D_, 2 * D_, fuF, nullptr, 0, 0, loF, goF);

  // 7) h2 = LN(h + fused)
  add_ln_kernel<<<M_, b256, 0, stream>>>(hf, fuF, n1_g, n1_b, h2F, h2B);

  // 8) FFN
  gemm_wmma<EPI_RELUBF><<<g1024, b32, 0, stream>>>(h2B, f1T, ffn_b1, 2 * D_, D_, nullptr, midB, 2 * D_, 0, nullptr, nullptr);
  gemm_wmma<EPI_F32   ><<<g512, b32, 0, stream>>>(midB, f2T, ffn_b2, D_, 2 * D_, ffF, nullptr, 0, 0, nullptr, nullptr);

  // 9) h3 = LN(LN(h2 + ffn; n2); n3)
  add_ln2_kernel<<<M_, b256, 0, stream>>>(h2F, ffF, n2_g, n2_b, n3_g, n3_b, h3F);

  // 10) mean-pool over seq + final projection
  pool_kernel<<<(B_ * D_) / 256, b256, 0, stream>>>(h3F, plF);
  outproj_kernel<<<(B_ * DOUT_) / 256, b256, 0, stream>>>(plF, out_w, out_b, (float*)d_out);
}
